// Tracking_17085379904335
// MI455X (gfx1250) — matrix operand order repaired and checked by reading, not compiled, not run on hardware
//
#include <hip/hip_runtime.h>
#include <cstdint>
#include <cstddef>

// ---------------- constants (match reference) ----------------
constexpr int   NBATCH  = 2;
constexpr int   NPTS    = 2048;
constexpr int   NCH     = 512;
constexpr int   KMAXS   = 12;     // storage stride for all knn index arrays
constexpr int   NB_ITER = 5;
constexpr int   CANDS   = 4;
constexpr int   SIMK    = 8;      // _matching similarity neighbors
constexpr int   OUTK    = 8;
constexpr float THRSIM  = 2.0f;
constexpr float THROUT  = 2.0f;
constexpr float EPSF    = 1e-8f;

typedef _Float16 half_t;
typedef __attribute__((ext_vector_type(16))) _Float16 v16h;
typedef __attribute__((ext_vector_type(8)))  _Float16 v8h;
typedef __attribute__((ext_vector_type(8)))  float    v8f;

__device__ __forceinline__ void split_bi(int t, int& b, int& i) { b = t / NPTS; i = t - b * NPTS; }

// ---------------- feature normalize -> f16 ----------------
__global__ void norm_feat_kernel(const float* __restrict__ f, half_t* __restrict__ out) {
    int row = blockIdx.x;                       // 0 .. NBATCH*NPTS-1
    const float* fr = f + (size_t)row * NCH;
    __shared__ float red[256];
    float s = 0.f;
    for (int c = threadIdx.x; c < NCH; c += 256) { float x = fr[c]; s += x * x; }
    red[threadIdx.x] = s;
    __syncthreads();
    for (int st = 128; st > 0; st >>= 1) {
        if ((int)threadIdx.x < st) red[threadIdx.x] += red[threadIdx.x + st];
        __syncthreads();
    }
    float inv = 1.0f / (sqrtf(red[0]) + EPSF);
    half_t* orow = out + (size_t)row * NCH;
    for (int c = threadIdx.x; c < NCH; c += 256) orow[c] = (half_t)(fr[c] * inv);
}

// ---------------- WMMA GEMM: S[b] = f1n[b] * f2n[b]^T ----------------
// grid (NPTS/64, NPTS/64, NBATCH), block 128 (4 waves; each wave owns a 32x32 tile)
__global__ __launch_bounds__(128)
void gemm_sim_kernel(const half_t* __restrict__ A, const half_t* __restrict__ Bm,
                     float* __restrict__ S) {
    const int bm = blockIdx.x, bn = blockIdx.y, bz = blockIdx.z;
    const int wave = threadIdx.x >> 5;
    const int lane = threadIdx.x & 31;
    const int wm = (wave >> 1) * 32, wn = (wave & 1) * 32;
    const half_t* Ab = A  + (size_t)bz * NPTS * NCH;
    const half_t* Bb = Bm + (size_t)bz * NPTS * NCH;
    const int l15 = lane & 15;
    const int hi  = lane >> 4;                 // half-wave id
    const int rA  = bm * 64 + wm + l15;        // A row (M)
    const int rB  = bn * 64 + wn + l15;        // B "column" = f2 row (N)
    v8f acc00 = {}, acc01 = {}, acc10 = {}, acc11 = {};
    for (int kk = 0; kk < NCH; kk += 32) {
        union { v16h v; v8h h[2]; } a0, a1, b0, b1;
        // A frag: lanes<16 hold K kk..kk+15 of row rA; lanes>=16 hold kk+16..kk+31
        const half_t* pa0 = Ab + (size_t)rA * NCH + kk + hi * 16;
        const half_t* pa1 = pa0 + 16 * NCH;
        a0.v = *(const v16h*)(pa0);
        a1.v = *(const v16h*)(pa1);
        // B frag: lanes<16 hold K kk..kk+15 of column rB; lanes>=16 hold kk+16..kk+31
        const half_t* pb0 = Bb + (size_t)rB * NCH + kk + hi * 16;
        const half_t* pb1 = pb0 + 16 * NCH;
        b0.v = *(const v16h*)(pb0);
        b1.v = *(const v16h*)(pb1);
        __builtin_prefetch(pa0 + 32, 0, 1);    // -> global_prefetch_b8
        __builtin_prefetch(pb0 + 32, 0, 1);
        acc00 = __builtin_amdgcn_wmma_f32_16x16x32_f16(false, a0.v, false, b0.v, (short)0, acc00, false, false);
        acc01 = __builtin_amdgcn_wmma_f32_16x16x32_f16(false, a0.v, false, b1.v, (short)0, acc01, false, false);
        acc10 = __builtin_amdgcn_wmma_f32_16x16x32_f16(false, a1.v, false, b0.v, (short)0, acc10, false, false);
        acc11 = __builtin_amdgcn_wmma_f32_16x16x32_f16(false, a1.v, false, b1.v, (short)0, acc11, false, false);
    }
    // C/D layout: VGPR r, lanes<16 -> M=r, lanes>=16 -> M=r+8; N = lane&15
    float* Sb = S + (size_t)bz * NPTS * NPTS;
    const int rowadd = (lane < 16) ? 0 : 8;
    for (int r = 0; r < 8; ++r) {
        int row0 = bm * 64 + wm + r + rowadd;
        int col0 = bn * 64 + wn + l15;
        Sb[(size_t)(row0)*NPTS + col0]           = acc00[r];
        Sb[(size_t)(row0)*NPTS + col0 + 16]      = acc01[r];
        Sb[(size_t)(row0 + 16)*NPTS + col0]      = acc10[r];
        Sb[(size_t)(row0 + 16)*NPTS + col0 + 16] = acc11[r];
    }
}

// ---------------- generic helpers ----------------
__global__ void zero_f_kernel(float* p, int n)            { int t = blockIdx.x*blockDim.x+threadIdx.x; if (t < n) p[t] = 0.f; }
__global__ void zero_i_kernel(int* p, int n)              { int t = blockIdx.x*blockDim.x+threadIdx.x; if (t < n) p[t] = 0; }
__global__ void fill_f_kernel(float* p, int n, float v)   { int t = blockIdx.x*blockDim.x+threadIdx.x; if (t < n) p[t] = v; }

__global__ void add_flow_kernel(const float* __restrict__ xy1, const float* __restrict__ fp,
                                float* __restrict__ out, int useflow) {
    int t = blockIdx.x*blockDim.x+threadIdx.x;
    if (t >= NBATCH*NPTS*2) return;
    out[t] = xy1[t] + (useflow ? fp[t] : 0.f);
}

// ---------------- unmasked KNN (squared dists, lowest-index ties) ----------------
__global__ void knn_kernel(const float* __restrict__ ref, const float* __restrict__ qry,
                           int* __restrict__ out, int k) {
    int t = blockIdx.x*blockDim.x+threadIdx.x;
    if (t >= NBATCH*NPTS) return;
    int b, i; split_bi(t, b, i); (void)i;
    const float* rxy = ref + (size_t)b * NPTS * 2;
    float qx = qry[(size_t)t*2], qy = qry[(size_t)t*2+1];
    float bd[KMAXS]; int bi_[KMAXS];
    for (int s = 0; s < k; ++s) { bd[s] = 3.0e38f; bi_[s] = 0; }
    for (int n = 0; n < NPTS; ++n) {
        float dx = qx - rxy[n*2], dy = qy - rxy[n*2+1];
        float d = dx*dx + dy*dy;
        if (d < bd[k-1]) {
            int p = k - 1;
            while (p > 0 && d < bd[p-1]) { bd[p] = bd[p-1]; bi_[p] = bi_[p-1]; --p; }
            bd[p] = d; bi_[p] = n;
        }
    }
    for (int s = 0; s < k; ++s) out[(size_t)t*KMAXS + s] = bi_[s];
}

// ---------------- sparse Sinkhorn ----------------
__device__ __forceinline__ float sink_eps(const float* epsilon) { return expf(epsilon[0]) + 0.03f; }
__device__ __forceinline__ float sink_pow(const float* gamma, const float* epsilon) {
    float e = sink_eps(epsilon), g = expf(gamma[0]);
    return g / (g + e);
}

__global__ void sink_u_kernel(const float* __restrict__ S, const int* __restrict__ cand,
                              const float* __restrict__ v, float* __restrict__ u,
                              const float* gamma, const float* epsilon, int dr0) {
    int t = blockIdx.x*blockDim.x+threadIdx.x;
    if (t >= NBATCH*NPTS) return;
    int b, i; split_bi(t, b, i);
    float eps = sink_eps(epsilon), power = sink_pow(gamma, epsilon);
    const float* Srow = S + ((size_t)b * NPTS + i) * NPTS;
    float s = 0.f;
    for (int q = 0; q < dr0; ++q) {
        int j = cand[(size_t)t*KMAXS + q];
        float K = expf(-(1.0f - Srow[j]) / eps);
        s += K * v[(size_t)b*NPTS + j];
    }
    u[t] = powf((1.0f / NPTS) / (s + EPSF), power);
}

__global__ void sink_scatter_kernel(const float* __restrict__ S, const int* __restrict__ cand,
                                    const float* __restrict__ u, float* __restrict__ colsum,
                                    const float* epsilon, int dr0) {
    int t = blockIdx.x*blockDim.x+threadIdx.x;
    if (t >= NBATCH*NPTS) return;
    int b, i; split_bi(t, b, i);
    float eps = sink_eps(epsilon);
    const float* Srow = S + ((size_t)b * NPTS + i) * NPTS;
    float ui = u[t];
    for (int q = 0; q < dr0; ++q) {
        int j = cand[(size_t)t*KMAXS + q];
        float K = expf(-(1.0f - Srow[j]) / eps);
        atomicAdd(&colsum[(size_t)b*NPTS + j], K * ui);
    }
}

__global__ void sink_v_kernel(const float* __restrict__ colsum, float* __restrict__ v,
                              const float* gamma, const float* epsilon) {
    int t = blockIdx.x*blockDim.x+threadIdx.x;
    if (t >= NBATCH*NPTS) return;
    float power = sink_pow(gamma, epsilon);
    v[t] = powf((1.0f / NPTS) / (colsum[t] + EPSF), power);
}

// ---------------- transport-plan mutual-best match (sparse) ----------------
__global__ void t_row_kernel(const float* __restrict__ S, const int* __restrict__ cand,
                             const float* __restrict__ u, const float* __restrict__ v,
                             float* __restrict__ rowv1, float* __restrict__ rowv2, int* __restrict__ rowi1,
                             const float* epsilon, int dr0) {
    int t = blockIdx.x*blockDim.x+threadIdx.x;
    if (t >= NBATCH*NPTS) return;
    int b, i; split_bi(t, b, i);
    float eps = sink_eps(epsilon);
    const float* Srow = S + ((size_t)b * NPTS + i) * NPTS;
    float ui = u[t];
    float b1 = -1.f, b2 = -1.f; int i1 = 0;
    for (int q = 0; q < dr0; ++q) {
        int j = cand[(size_t)t*KMAXS + q];
        float K = expf(-(1.0f - Srow[j]) / eps);
        float tv = ui * K * v[(size_t)b*NPTS + j];
        if (tv > b1) { b2 = b1; b1 = tv; i1 = j; }
        else if (tv > b2) { b2 = tv; }
    }
    if (b1 < 0.f) { b1 = 0.f; i1 = 0; }
    rowv1[t] = b1; rowv2[t] = fmaxf(b2, 0.f); rowi1[t] = i1;    // row has implicit zeros
}

__global__ void t_colzero_kernel(unsigned long long* colenc) {
    int t = blockIdx.x*blockDim.x+threadIdx.x;
    if (t >= NBATCH*NPTS) return;
    colenc[t] = 0x7FFFFFFFull;   // value 0 at row 0 (jax lowest-index tiebreak)
}

__global__ void t_colscatter_kernel(const float* __restrict__ S, const int* __restrict__ cand,
                                    const float* __restrict__ u, const float* __restrict__ v,
                                    unsigned long long* __restrict__ colenc,
                                    const float* epsilon, int dr0) {
    int t = blockIdx.x*blockDim.x+threadIdx.x;
    if (t >= NBATCH*NPTS) return;
    int b, i; split_bi(t, b, i);
    float eps = sink_eps(epsilon);
    const float* Srow = S + ((size_t)b * NPTS + i) * NPTS;
    float ui = u[t];
    for (int q = 0; q < dr0; ++q) {
        int j = cand[(size_t)t*KMAXS + q];
        float K = expf(-(1.0f - Srow[j]) / eps);
        float tv = ui * K * v[(size_t)b*NPTS + j];
        unsigned long long e = ((unsigned long long)__float_as_uint(tv) << 32)
                             | (unsigned)(0x7FFFFFFF - i);
        atomicMax(&colenc[(size_t)b*NPTS + j], e);
    }
}

__global__ void recon_kernel(const float* __restrict__ rowv1, const float* __restrict__ rowv2,
                             const int* __restrict__ rowi1, const unsigned long long* __restrict__ colenc,
                             int* __restrict__ idx) {
    int t = blockIdx.x*blockDim.x+threadIdx.x;
    if (t >= NBATCH*NPTS) return;
    int b, i; split_bi(t, b, i);
    int bj = rowi1[t];
    int back = 0x7FFFFFFF - (int)(colenc[(size_t)b*NPTS + bj] & 0x7FFFFFFFull);
    bool mutual = (back == i);
    bool ok = rowv1[t] > 1.2f * rowv2[t];
    idx[t] = (mutual && ok) ? bj : -1;
}

// ---------------- dense sqrt-distance matcher (knn_match path) ----------------
__global__ void dense_row_kernel(const float* __restrict__ xy11, const float* __restrict__ xy2,
                                 float* __restrict__ rowv1, float* __restrict__ rowv2, int* __restrict__ rowi1) {
    int t = blockIdx.x*blockDim.x+threadIdx.x;
    if (t >= NBATCH*NPTS) return;
    int b, i; split_bi(t, b, i); (void)i;
    const float* p2 = xy2 + (size_t)b * NPTS * 2;
    float qx = xy11[(size_t)t*2], qy = xy11[(size_t)t*2+1];
    float b1 = 3.0e38f, b2 = 3.0e38f; int i1 = 0;
    for (int j = 0; j < NPTS; ++j) {
        float dx = qx - p2[j*2], dy = qy - p2[j*2+1];
        float d = sqrtf(dx*dx + dy*dy + EPSF);
        if (d < b1) { b2 = b1; b1 = d; i1 = j; }
        else if (d < b2) { b2 = d; }
    }
    rowv1[t] = b1; rowv2[t] = b2; rowi1[t] = i1;
}

__global__ void dense_col_kernel(const float* __restrict__ xy11, const float* __restrict__ xy2,
                                 int* __restrict__ colarg) {
    int t = blockIdx.x*blockDim.x+threadIdx.x;
    if (t >= NBATCH*NPTS) return;
    int b, j; split_bi(t, b, j);
    const float* p1 = xy11 + (size_t)b * NPTS * 2;
    float cx = xy2[(size_t)t*2], cy = xy2[(size_t)t*2+1];
    float best = 3.0e38f; int bi_ = 0;
    for (int i = 0; i < NPTS; ++i) {
        float dx = p1[i*2] - cx, dy = p1[i*2+1] - cy;
        float d = sqrtf(dx*dx + dy*dy + EPSF);
        if (d < best) { best = d; bi_ = i; }
    }
    colarg[t] = bi_;
}

__global__ void recon_min_kernel(const float* __restrict__ rowv1, const float* __restrict__ rowv2,
                                 const int* __restrict__ rowi1, const int* __restrict__ colarg,
                                 int* __restrict__ idx) {
    int t = blockIdx.x*blockDim.x+threadIdx.x;
    if (t >= NBATCH*NPTS) return;
    int b, i; split_bi(t, b, i);
    int bj = rowi1[t];
    bool mutual = (colarg[(size_t)b*NPTS + bj] == i);
    bool ok = rowv1[t] * 1.2f < rowv2[t];
    idx[t] = (mutual && ok) ? bj : -1;
}

// ---------------- filters ----------------
__global__ void appear_kernel(int* __restrict__ idx, const int* __restrict__ cand, int dr0) {
    int t = blockIdx.x*blockDim.x+threadIdx.x;
    if (t >= NBATCH*NPTS) return;
    int v = idx[t];
    bool ok = false;
    for (int q = 0; q < dr0; ++q) ok = ok || (cand[(size_t)t*KMAXS + q] == v);
    idx[t] = ok ? v : -1;
}

__global__ void sim_kernel(const float* __restrict__ xy1, const float* __restrict__ xy2,
                           int* __restrict__ idx, const int* __restrict__ nb0, const int* __restrict__ nb1,
                           int knb, float thr) {
    int t = blockIdx.x*blockDim.x+threadIdx.x;
    if (t >= NBATCH*NPTS) return;
    int b, i; split_bi(t, b, i); (void)i;
    int v = idx[t];
    int j = (v < 0) ? 0 : v;
    float o1x[11], o1y[11], o2x[11], o2y[11];
    float cx = xy1[(size_t)t*2], cy = xy1[(size_t)t*2+1];
    for (int q = 0; q < knb; ++q) {
        int n = nb0[(size_t)t*KMAXS + 1 + q];
        o1x[q] = xy1[((size_t)b*NPTS + n)*2]   - cx;
        o1y[q] = xy1[((size_t)b*NPTS + n)*2+1] - cy;
    }
    size_t jt = (size_t)b*NPTS + j;
    float jx = xy2[jt*2], jy = xy2[jt*2+1];
    for (int q = 0; q < knb; ++q) {
        int n = nb1[jt*KMAXS + 1 + q];
        o2x[q] = xy2[((size_t)b*NPTS + n)*2]   - jx;
        o2y[q] = xy2[((size_t)b*NPTS + n)*2+1] - jy;
    }
    float s1 = 0.f;
    for (int a = 0; a < knb; ++a) {
        float mn = 3.0e38f;
        for (int c = 0; c < knb; ++c) {
            float dx = o1x[a]-o2x[c], dy = o1y[a]-o2y[c];
            mn = fminf(mn, sqrtf(dx*dx + dy*dy + EPSF));
        }
        s1 += mn;
    }
    float s2 = 0.f;
    for (int c = 0; c < knb; ++c) {
        float mn = 3.0e38f;
        for (int a = 0; a < knb; ++a) {
            float dx = o1x[a]-o2x[c], dy = o1y[a]-o2y[c];
            mn = fminf(mn, sqrtf(dx*dx + dy*dy + EPSF));
        }
        s2 += mn;
    }
    float cham = 0.5f * (s1 / knb + s2 / knb);
    idx[t] = (v >= 0 && cham <= thr) ? v : -1;
}

__global__ void flowmask_kernel(const float* __restrict__ xy1, const float* __restrict__ xy2,
                                const int* __restrict__ idx, float* __restrict__ flow, int* __restrict__ mask) {
    int t = blockIdx.x*blockDim.x+threadIdx.x;
    if (t >= NBATCH*NPTS) return;
    int b, i; split_bi(t, b, i); (void)i;
    int v = idx[t];
    int m = (v >= 0) ? 1 : 0;
    int j = m ? v : 0;
    size_t jt = (size_t)b*NPTS + j;
    flow[(size_t)t*2]   = m ? (xy2[jt*2]   - xy1[(size_t)t*2])   : 0.f;
    flow[(size_t)t*2+1] = m ? (xy2[jt*2+1] - xy1[(size_t)t*2+1]) : 0.f;
    mask[t] = m;
}

__global__ void outlier_kernel(const float* __restrict__ xy1, const int* __restrict__ mask,
                               const float* __restrict__ flow, const int* __restrict__ idx_in,
                               int* __restrict__ idx1_out, float thr) {
    int t = blockIdx.x*blockDim.x+threadIdx.x;
    if (t >= NBATCH*NPTS) return;
    int b, i; split_bi(t, b, i);
    const float* pb = xy1 + (size_t)b * NPTS * 2;
    float qx = pb[i*2], qy = pb[i*2+1];
    float bd[OUTK]; int bi_[OUTK];
    for (int s = 0; s < OUTK; ++s) { bd[s] = 3.0e38f; bi_[s] = -1; }
    int found = 0;
    for (int n = 0; n < NPTS; ++n) {
        if (n == i) continue;
        if (!mask[(size_t)b*NPTS + n]) continue;
        float dx = qx - pb[n*2], dy = qy - pb[n*2+1];
        float d = dx*dx + dy*dy;
        ++found;
        if (d < bd[OUTK-1]) {
            int p = OUTK - 1;
            while (p > 0 && d < bd[p-1]) { bd[p] = bd[p-1]; bi_[p] = bi_[p-1]; --p; }
            bd[p] = d; bi_[p] = n;
        }
    }
    int cnt = found < OUTK ? found : OUTK;
    float mx = 0.f, my = 0.f;
    for (int s = 0; s < cnt; ++s) {
        size_t nt = (size_t)b*NPTS + bi_[s];
        mx += flow[nt*2]; my += flow[nt*2+1];
    }
    float den = (cnt > 0) ? (float)cnt : 1.f;
    mx /= den; my /= den;
    float fx = flow[(size_t)t*2], fy = flow[(size_t)t*2+1];
    float dev = sqrtf((fx-mx)*(fx-mx) + (fy-my)*(fy-my) + EPSF);
    int m = mask[t];
    bool keep = m && ((cnt > 0) ? (dev <= thr) : true);
    idx1_out[t] = keep ? (idx_in[t] + 1) : 0;
}

__global__ void merge_kernel(const int* __restrict__ idx1, int* __restrict__ idx0) {
    int t = blockIdx.x*blockDim.x+threadIdx.x;
    if (t >= NBATCH*NPTS) return;
    int p = idx1[t];
    if (p != 0) idx0[t] = p - 1;   // keep previous match; else keep new idx0
}

__global__ void dedup_count_kernel(const int* __restrict__ idx1, int* __restrict__ counts) {
    int t = blockIdx.x*blockDim.x+threadIdx.x;
    if (t >= NBATCH*NPTS) return;
    int b, i; split_bi(t, b, i); (void)i;
    atomicAdd(&counts[(size_t)b*(NPTS+1) + idx1[t]], 1);
}

__global__ void dedup_apply_kernel(int* __restrict__ idx1, const int* __restrict__ counts) {
    int t = blockIdx.x*blockDim.x+threadIdx.x;
    if (t >= NBATCH*NPTS) return;
    int b, i; split_bi(t, b, i); (void)i;
    int v = idx1[t];
    if (counts[(size_t)b*(NPTS+1) + v] > 1) idx1[t] = 0;
}

// ---------------- finish: flow + inverse-distance interpolation ----------------
__global__ void finish_flow_kernel(const float* __restrict__ xy1, const float* __restrict__ xy2,
                                   const int* __restrict__ idx1, float* __restrict__ flow, int* __restrict__ mask) {
    int t = blockIdx.x*blockDim.x+threadIdx.x;
    if (t >= NBATCH*NPTS) return;
    int b, i; split_bi(t, b, i); (void)i;
    int v = idx1[t];
    int m = (v != 0) ? 1 : 0;
    int j = m ? (v - 1) : 0;
    size_t jt = (size_t)b*NPTS + j;
    flow[(size_t)t*2]   = m ? (xy2[jt*2]   - xy1[(size_t)t*2])   : 0.f;
    flow[(size_t)t*2+1] = m ? (xy2[jt*2+1] - xy1[(size_t)t*2+1]) : 0.f;
    mask[t] = m;
}

__global__ void griddata_kernel(const float* __restrict__ xy1, const float* __restrict__ flow,
                                const int* __restrict__ mask, float* __restrict__ flowg) {
    int t = blockIdx.x*blockDim.x+threadIdx.x;
    if (t >= NBATCH*NPTS) return;
    int b, i; split_bi(t, b, i);
    if (mask[t]) {
        flowg[(size_t)t*2]   = flow[(size_t)t*2];
        flowg[(size_t)t*2+1] = flow[(size_t)t*2+1];
        return;
    }
    const float* pb = xy1 + (size_t)b * NPTS * 2;
    float qx = pb[i*2], qy = pb[i*2+1];
    const int GK = 4;
    float bd[GK]; int bi_[GK];
    for (int s = 0; s < GK; ++s) { bd[s] = 3.0e38f; bi_[s] = -1; }
    int found = 0;
    for (int n = 0; n < NPTS; ++n) {
        if (!mask[(size_t)b*NPTS + n]) continue;
        float dx = qx - pb[n*2], dy = qy - pb[n*2+1];
        float d = dx*dx + dy*dy;           // squared, like reference weights
        ++found;
        if (d < bd[GK-1]) {
            int p = GK - 1;
            while (p > 0 && d < bd[p-1]) { bd[p] = bd[p-1]; bi_[p] = bi_[p-1]; --p; }
            bd[p] = d; bi_[p] = n;
        }
    }
    int cnt = found < GK ? found : GK;
    float wsum = 0.f, ix = 0.f, iy = 0.f;
    for (int s = 0; s < cnt; ++s) {
        float w = 1.0f / (bd[s] + 1e-6f);
        size_t nt = (size_t)b*NPTS + bi_[s];
        ix += w * flow[nt*2]; iy += w * flow[nt*2+1];
        wsum += w;
    }
    flowg[(size_t)t*2]   = ix / (wsum + EPSF);
    flowg[(size_t)t*2+1] = iy / (wsum + EPSF);
}

__global__ void out_kernel(const float* __restrict__ flow, const int* __restrict__ mask,
                           float* __restrict__ out) {
    int t = blockIdx.x*blockDim.x+threadIdx.x;
    if (t >= NBATCH*NPTS) return;
    out[(size_t)t*2]   = flow[(size_t)t*2];
    out[(size_t)t*2+1] = flow[(size_t)t*2+1];
    out[(size_t)NBATCH*NPTS*2 + t] = (float)mask[t];
}

// ---------------- host orchestration ----------------
extern "C" void kernel_launch(void* const* d_in, const int* in_sizes, int n_in,
                              void* d_out, int out_size, void* d_ws, size_t ws_size,
                              hipStream_t stream) {
    (void)in_sizes; (void)n_in; (void)out_size; (void)ws_size;
    const float* xy1     = (const float*)d_in[0];
    const float* xy2     = (const float*)d_in[1];
    const float* f1      = (const float*)d_in[2];
    const float* f2      = (const float*)d_in[3];
    const float* gamma   = (const float*)d_in[4];
    const float* epsilon = (const float*)d_in[5];
    float* out = (float*)d_out;

    // bump allocator over d_ws (256B aligned)
    char* wp = (char*)d_ws;
    auto alloc = [&](size_t bytes) -> void* {
        void* p = (void*)wp;
        wp += (bytes + 255) & ~(size_t)255;
        return p;
    };
    const size_t BN = (size_t)NBATCH * NPTS;
    half_t* f1h = (half_t*)alloc(BN * NCH * sizeof(half_t));
    half_t* f2h = (half_t*)alloc(BN * NCH * sizeof(half_t));
    float*  S   = (float*) alloc(BN * NPTS * sizeof(float));
    int* nb0    = (int*)   alloc(BN * KMAXS * sizeof(int));
    int* nb1    = (int*)   alloc(BN * KMAXS * sizeof(int));
    int* cand   = (int*)   alloc(BN * KMAXS * sizeof(int));
    float* u    = (float*) alloc(BN * sizeof(float));
    float* v    = (float*) alloc(BN * sizeof(float));
    float* colsum = (float*)alloc(BN * sizeof(float));
    float* rowv1 = (float*)alloc(BN * sizeof(float));
    float* rowv2 = (float*)alloc(BN * sizeof(float));
    int*   rowi1 = (int*)  alloc(BN * sizeof(int));
    unsigned long long* colenc = (unsigned long long*)alloc(BN * sizeof(unsigned long long));
    int*   colarg = (int*) alloc(BN * sizeof(int));
    int*   idx0   = (int*) alloc(BN * sizeof(int));
    int*   idx1   = (int*) alloc(BN * sizeof(int));
    int*   counts = (int*) alloc((size_t)NBATCH * (NPTS+1) * sizeof(int));
    float* flow   = (float*)alloc(BN * 2 * sizeof(float));
    float* flowg  = (float*)alloc(BN * 2 * sizeof(float));
    int*   maskA  = (int*) alloc(BN * sizeof(int));
    float* xy11   = (float*)alloc(BN * 2 * sizeof(float));

    const dim3 blk(256);
    const dim3 gPT((unsigned)((BN + 255) / 256));            // per-point kernels
    const dim3 gPT2((unsigned)((BN * 2 + 255) / 256));       // per-coordinate kernels
    const dim3 gCNT((unsigned)((NBATCH * (NPTS+1) + 255) / 256));

    // 1) normalized f16 features + similarity matrix via WMMA (shared by both matchings)
    norm_feat_kernel<<<dim3((unsigned)BN), blk, 0, stream>>>(f1, f1h);
    norm_feat_kernel<<<dim3((unsigned)BN), blk, 0, stream>>>(f2, f2h);
    gemm_sim_kernel<<<dim3(NPTS/64, NPTS/64, NBATCH), dim3(128), 0, stream>>>(f1h, f2h, S);

    auto sinkhorn_round = [&](int dr0) {
        fill_f_kernel<<<gPT, blk, 0, stream>>>(u, (int)BN, 1.0f);
        fill_f_kernel<<<gPT, blk, 0, stream>>>(v, (int)BN, 1.0f);
        for (int it = 0; it < NB_ITER; ++it) {
            sink_u_kernel<<<gPT, blk, 0, stream>>>(S, cand, v, u, gamma, epsilon, dr0);
            zero_f_kernel<<<gPT, blk, 0, stream>>>(colsum, (int)BN);
            sink_scatter_kernel<<<gPT, blk, 0, stream>>>(S, cand, u, colsum, epsilon, dr0);
            sink_v_kernel<<<gPT, blk, 0, stream>>>(colsum, v, gamma, epsilon);
        }
        t_row_kernel<<<gPT, blk, 0, stream>>>(S, cand, u, v, rowv1, rowv2, rowi1, epsilon, dr0);
        t_colzero_kernel<<<gPT, blk, 0, stream>>>(colenc);
        t_colscatter_kernel<<<gPT, blk, 0, stream>>>(S, cand, u, v, colenc, epsilon, dr0);
        recon_kernel<<<gPT, blk, 0, stream>>>(rowv1, rowv2, rowi1, colenc, idx0);
    };

    auto do_finish = [&](void) {
        finish_flow_kernel<<<gPT, blk, 0, stream>>>(xy1, xy2, idx1, flow, maskA);
        griddata_kernel<<<gPT, blk, 0, stream>>>(xy1, flow, maskA, flowg);
    };

    auto matching = [&](int useflow) {
        add_flow_kernel<<<gPT2, blk, 0, stream>>>(xy1, flowg, xy11, useflow);
        knn_kernel<<<gPT, blk, 0, stream>>>(xy1, xy1, nb0, SIMK + 1);
        knn_kernel<<<gPT, blk, 0, stream>>>(xy2, xy2, nb1, SIMK + 1);
        knn_kernel<<<gPT, blk, 0, stream>>>(xy2, xy11, cand, CANDS + 1);
        for (int it = 0; it < CANDS; ++it) {
            int dr0 = it + 1;
            sinkhorn_round(dr0);
            if (it > 0) merge_kernel<<<gPT, blk, 0, stream>>>(idx1, idx0);
            appear_kernel<<<gPT, blk, 0, stream>>>(idx0, cand, dr0);
            sim_kernel<<<gPT, blk, 0, stream>>>(xy1, xy2, idx0, nb0, nb1, SIMK, THRSIM);
            flowmask_kernel<<<gPT, blk, 0, stream>>>(xy1, xy2, idx0, flow, maskA);
            outlier_kernel<<<gPT, blk, 0, stream>>>(xy1, maskA, flow, idx0, idx1, THROUT);
            if (it > 0) {
                zero_i_kernel<<<gCNT, blk, 0, stream>>>(counts, NBATCH * (NPTS+1));
                dedup_count_kernel<<<gPT, blk, 0, stream>>>(idx1, counts);
                dedup_apply_kernel<<<gPT, blk, 0, stream>>>(idx1, counts);
            }
        }
        do_finish();
    };

    auto knn_match = [&](void) {
        add_flow_kernel<<<gPT2, blk, 0, stream>>>(xy1, flowg, xy11, 1);
        knn_kernel<<<gPT, blk, 0, stream>>>(xy1, xy1, nb0, 12);
        knn_kernel<<<gPT, blk, 0, stream>>>(xy2, xy2, nb1, 12);
        knn_kernel<<<gPT, blk, 0, stream>>>(xy2, xy11, cand, 12);
        dense_row_kernel<<<gPT, blk, 0, stream>>>(xy11, xy2, rowv1, rowv2, rowi1);
        dense_col_kernel<<<gPT, blk, 0, stream>>>(xy11, xy2, colarg);
        recon_min_kernel<<<gPT, blk, 0, stream>>>(rowv1, rowv2, rowi1, colarg, idx0);
        appear_kernel<<<gPT, blk, 0, stream>>>(idx0, cand, 1);
        sim_kernel<<<gPT, blk, 0, stream>>>(xy1, xy2, idx0, nb0, nb1, 11, THRSIM);
        flowmask_kernel<<<gPT, blk, 0, stream>>>(xy1, xy2, idx0, flow, maskA);
        outlier_kernel<<<gPT, blk, 0, stream>>>(xy1, maskA, flow, idx0, idx1, THROUT);
        do_finish();
    };

    matching(/*useflow=*/0);   // flow0 = 0
    matching(/*useflow=*/1);   // flow_pred = flow_gri
    knn_match();
    knn_match();

    out_kernel<<<gPT, blk, 0, stream>>>(flow, maskA, out);
}
